// DGTBlock_11261404250651
// MI455X (gfx1250) — matrix operand order repaired and checked by reading, not compiled, not run on hardware
//
#include <hip/hip_runtime.h>

#define DEVINL __device__ __forceinline__

// ---------------- problem sizes ----------------
constexpr int Bc  = 4;
constexpr int Nn  = 4096;
constexpr int KNN = 16;
constexpr int DP  = 64;
constexpr int DM  = 128;
constexpr float BN_EPS = 1e-5f;
constexpr float INV_T  = 0.015625f;  // 1/sqrt(N) = 1/64

typedef float    v8f   __attribute__((ext_vector_type(8)));
typedef __bf16   v16bf __attribute__((ext_vector_type(16)));
typedef __bf16   v8bf  __attribute__((ext_vector_type(8)));
typedef unsigned v4u   __attribute__((ext_vector_type(4)));
typedef unsigned v8u   __attribute__((ext_vector_type(8)));

#if __has_builtin(__builtin_amdgcn_wmma_f32_16x16x4_f32)
#define HAVE_WMMA_F32X4 1
typedef float v2f __attribute__((ext_vector_type(2)));
#endif

DEVINL float lrelu(float x) { return x >= 0.f ? x : 0.2f * x; }

DEVINL unsigned lds_off_u32(const void* p) {
  // generic pointers to LDS carry the byte offset in the low 32 bits (aperture scheme)
  return (unsigned)(unsigned long long)(uintptr_t)p;
}

// ---- Tensor Data Mover: 1-D bf16 tile, global -> LDS (D# per ISA 08_async_tensor) ----
DEVINL void tdm_load_1d_bf16(const __bf16* gsrc, unsigned lds_byte_off, unsigned numel) {
  unsigned long long ga = (unsigned long long)(uintptr_t)gsrc;
  v4u g0;
  g0[0] = 1u;                                                  // count=1, user descriptor
  g0[1] = lds_byte_off;                                        // lds_addr
  g0[2] = (unsigned)(ga & 0xFFFFFFFFull);                      // global_addr[31:0]
  g0[3] = (unsigned)((ga >> 32) & 0x1FFFFFFull) | (2u << 30);  // global_addr[56:32] | type=2
  v8u g1;
  g1[0] = 0x00010000u;                 // workgroup_mask=0, data_size=1 (2 bytes)
  g1[1] = (numel & 0xFFFFu) << 16;     // tensor_dim0[15:0]
  g1[2] = (numel >> 16) | (1u << 16);  // tensor_dim0[31:16] | tensor_dim1=1
  g1[3] = (numel & 0xFFFFu) << 16;     // tile_dim0 = numel (1-D tile)
  g1[4] = 0u;                          // tile_dim1=0 (unused), tile_dim2=0
  g1[5] = numel;                       // tensor_dim0_stride lo
  g1[6] = 0u;
  g1[7] = 0u;
  v4u z = {0u, 0u, 0u, 0u};
  asm volatile("tensor_load_to_lds %0, %1, %2, %3"
               :: "s"(g0), "s"(g1), "s"(z), "s"(z)
               : "memory");
}

// ---------------- WMMA bf16 16x16x32 fragment loaders ----------------
// A (16x32): lanes 0-15 row M=lane, K=k0..k0+15;
// lanes 16-31 row M=lane-16, K=k0+16..k0+31.
DEVINL v16bf ldsA_frag(const __bf16* Wm, int ldw, int row, int k0, int lane) {
  const int base = k0 + ((lane & 16) ? 16 : 0);
  v8bf lo = *reinterpret_cast<const v8bf*>(Wm + row * ldw + base);
  v8bf hi = *reinterpret_cast<const v8bf*>(Wm + row * ldw + base + 8);
  return __builtin_shufflevector(lo, hi, 0,1,2,3,4,5,6,7,8,9,10,11,12,13,14,15);
}
// B (32x16): lanes 0-15 col N=lane, K=k0..k0+15; lanes 16-31 col N=lane-16, K=k0+16..31.
// Activations column-major -> one 32-byte load.
DEVINL v16bf ldsB_frag(const __bf16* act, int ldc, int col, int k0, int lane) {
  const int off = col * ldc + k0 + ((lane & 16) ? 16 : 0);
  return *reinterpret_cast<const v16bf*>(act + off);
}
DEVINL v16bf gblB_frag_bf16T(const __bf16* X, int col, int k0, int lane) {
  const int kb = k0 + ((lane & 16) ? 16 : 0);
  return *reinterpret_cast<const v16bf*>(X + (size_t)col * DM + kb);
}
DEVINL v16bf gblA_frag_f32(const float* W, int ldw, int row, int k0, int lane) {
  v16bf a;
  const int base = k0 + ((lane & 16) ? 16 : 0);
#pragma unroll
  for (int e = 0; e < 8; ++e) {
    a[e]     = (__bf16)W[row * ldw + base + e];
    a[e + 8] = (__bf16)W[row * ldw + base + 8 + e];
  }
  return a;
}
DEVINL v16bf gblB_frag_f32(const float* X, int ldx, int col, int k0, int lane) {
  v16bf b;
  const int kb = k0 + ((lane & 16) ? 16 : 0);
#pragma unroll
  for (int e = 0; e < 16; ++e) b[e] = (__bf16)X[(size_t)(kb + e) * ldx + col];
  return b;
}

DEVINL void bn_stage(const float* p, int C, int c, float* sc, float* sh) {
  float g = p[c], be = p[C + c], m = p[2 * C + c], va = p[3 * C + c];
  float s = g * rsqrtf(va + BN_EPS);
  sc[c] = s;
  sh[c] = be - m * s;
}

// GEMM (DM x DM) * (DM x nCols) from LDS, bn+lrelu epilogue, column-major act in/out.
DEVINL void gemm_lds_bn_lrelu(const __bf16* sWm, const __bf16* sIn, __bf16* sOut,
                              const float* sc_p, const float* sh_p,
                              int nCols, int lane, int wv) {
  const int m0    = wv * 16;
  const int rowA  = m0 + (lane & 15);
  const int rbase = m0 + ((lane & 16) ? 8 : 0);
  float sc[8], sh[8];
#pragma unroll
  for (int v = 0; v < 8; ++v) { sc[v] = sc_p[rbase + v]; sh[v] = sh_p[rbase + v]; }
  for (int t = 0; t < nCols / 16; ++t) {
    v8f acc = {};
#pragma unroll
    for (int kk = 0; kk < DM / 32; ++kk) {
      v16bf af = ldsA_frag(sWm, DM, rowA, kk * 32, lane);
      v16bf bf = ldsB_frag(sIn, DM, t * 16 + (lane & 15), kk * 32, lane);
      acc = __builtin_amdgcn_wmma_f32_16x16x32_bf16(false, af, false, bf,
                                                    (short)0, acc, false, false);
    }
    const int col = t * 16 + (lane & 15);
#pragma unroll
    for (int v = 0; v < 8; ++v) {
      float y = acc[v] * sc[v] + sh[v];
      sOut[col * DM + rbase + v] = (__bf16)lrelu(y);
    }
  }
}

// ---------------- kernel 0: bf16 weight cache [Wd2 | Wg1 | Wg2 | W2] ----------------
__global__ __launch_bounds__(256) void prep_weights_kernel(const float* __restrict__ Wd2,
                                                           const float* __restrict__ Wg1,
                                                           const float* __restrict__ Wg2,
                                                           const float* __restrict__ W2f,
                                                           __bf16* __restrict__ wout) {
  int i = blockIdx.x * 256 + threadIdx.x;   // < 3*DM*DM + DP*DM
  float v;
  if (i < DM * DM)              v = Wd2[i];
  else if (i < 2 * DM * DM)     v = Wg1[i - DM * DM];
  else if (i < 3 * DM * DM)     v = Wg2[i - 2 * DM * DM];
  else                          v = W2f[i - 3 * DM * DM];
  wout[i] = (__bf16)v;
}

// ---------------- kernel 1: squared norms ----------------
__global__ __launch_bounds__(256) void sq_kernel(const float* __restrict__ feats,
                                                 float* __restrict__ sq) {
  int g = blockIdx.x * 256 + threadIdx.x;  // 0 .. B*N-1
  int b = g / Nn, n = g % Nn;
  const float* F = feats + (size_t)b * DP * Nn;
  float s = 0.f;
#pragma unroll
  for (int c = 0; c < DP; ++c) { float x = F[(size_t)c * Nn + n]; s += x * x; }
  sq[g] = s;
}

// ---------------- kernel 2: kNN via WMMA Gram tiles + register top-16 ----------------
__global__ __launch_bounds__(256) void knn_kernel(const float* __restrict__ feats,
                                                  const float* __restrict__ sq,
                                                  int* __restrict__ idxOut) {
  const int tid = threadIdx.x, lane = tid & 31, wv = tid >> 5;
  const int b = blockIdx.y;
  const int r0 = blockIdx.x * 128 + wv * 16;
  const float* F = feats + (size_t)b * DP * Nn;
  __shared__ float sD[8][16][16];

#ifdef HAVE_WMMA_F32X4
  v2f aF[16];
  {
    const int r = r0 + (lane & 15);
#pragma unroll
    for (int kk = 0; kk < 16; ++kk) {
      const int c0 = 4 * kk + ((lane & 16) ? 2 : 0);
      aF[kk][0] = F[(size_t)c0 * Nn + r];
      aF[kk][1] = F[(size_t)(c0 + 1) * Nn + r];
    }
  }
#else
  v16bf aB[2];
  {
    const int r = r0 + (lane & 15);
#pragma unroll
    for (int kk = 0; kk < 2; ++kk) {
      const int base = kk * 32 + ((lane & 16) ? 16 : 0);
#pragma unroll
      for (int e = 0; e < 8; ++e) {
        aB[kk][e]     = (__bf16)F[(size_t)(base + e) * Nn + r];
        aB[kk][e + 8] = (__bf16)F[(size_t)(base + 8 + e) * Nn + r];
      }
    }
  }
#endif

  float bd[16]; int bi[16];
#pragma unroll
  for (int s = 0; s < 16; ++s) { bd[s] = 3.0e38f; bi[s] = 0; }

  float sqRow[8];
  const int rbase = r0 + ((lane & 16) ? 8 : 0);
#pragma unroll
  for (int v = 0; v < 8; ++v) sqRow[v] = sq[(size_t)b * Nn + rbase + v];

  for (int t = 0; t < Nn / 16; ++t) {
    const int m = t * 16 + (lane & 15);
    v8f acc = {};
#ifdef HAVE_WMMA_F32X4
#pragma unroll
    for (int kk = 0; kk < 16; ++kk) {
      const int c0 = 4 * kk + ((lane & 16) ? 2 : 0);
      v2f bF;
      bF[0] = F[(size_t)c0 * Nn + m];
      bF[1] = F[(size_t)(c0 + 1) * Nn + m];
      acc = __builtin_amdgcn_wmma_f32_16x16x4_f32(false, aF[kk], false, bF,
                                                  (short)0, acc, false, false);
    }
#else
#pragma unroll
    for (int kk = 0; kk < 2; ++kk) {
      v16bf bB = gblB_frag_f32(F, Nn, m, kk * 32, lane);
      acc = __builtin_amdgcn_wmma_f32_16x16x32_bf16(false, aB[kk], false, bB,
                                                    (short)0, acc, false, false);
    }
#endif
    const float sqm = sq[(size_t)b * Nn + m];
#pragma unroll
    for (int v = 0; v < 8; ++v) {
      const int rloc = ((lane & 16) ? 8 : 0) + v;
      sD[wv][rloc][lane & 15] = sqRow[v] + sqm - 2.0f * acc[v];
    }
    if (lane < 16) {
#pragma unroll
      for (int jj = 0; jj < 16; ++jj) {
        float d = sD[wv][lane][jj];
        int  mi = t * 16 + jj;
        if (d < bd[15]) {
          bool placed = false;
#pragma unroll
          for (int s = 15; s >= 1; --s) {
            if (!placed) {
              if (d < bd[s - 1]) { bd[s] = bd[s - 1]; bi[s] = bi[s - 1]; }
              else { bd[s] = d; bi[s] = mi; placed = true; }
            }
          }
          if (!placed) { bd[0] = d; bi[0] = mi; }
        }
      }
    }
  }
  if (lane < 16) {
    const size_t base = ((size_t)b * Nn + (r0 + lane)) * KNN;
#pragma unroll
    for (int s = 0; s < 16; ++s) idxOut[base + s] = bi[s];
  }
}

// ---------------- kernel 3: x = lrelu(bn1(W1 @ feats)), stored point-major [n][c] ----------------
__global__ __launch_bounds__(256) void compute_x_kernel(const float* __restrict__ feats,
                                                        const float* __restrict__ W1,
                                                        const float* __restrict__ bn1,
                                                        __bf16* __restrict__ x) {
  const int tid = threadIdx.x, lane = tid & 31, wv = tid >> 5;
  const int b = blockIdx.y;
  const int m0 = wv * 16;
  const int rowA = m0 + (lane & 15);
  const int rbase = m0 + ((lane & 16) ? 8 : 0);
  const int col = blockIdx.x * 16 + (lane & 15);
  const float* F = feats + (size_t)b * DP * Nn;
  v8f acc = {};
#pragma unroll
  for (int kk = 0; kk < 2; ++kk) {
    v16bf af = gblA_frag_f32(W1, DP, rowA, kk * 32, lane);
    v16bf bf = gblB_frag_f32(F, Nn, col, kk * 32, lane);
    acc = __builtin_amdgcn_wmma_f32_16x16x32_bf16(false, af, false, bf,
                                                  (short)0, acc, false, false);
  }
#pragma unroll
  for (int v = 0; v < 8; ++v) {
    const int r = rbase + v;
    float s = bn1[r] * rsqrtf(bn1[3 * DM + r] + BN_EPS);
    float y = lrelu((acc[v] - bn1[2 * DM + r]) * s + bn1[DM + r]);
    x[(size_t)b * DM * Nn + (size_t)col * DM + r] = (__bf16)y;
  }
}

// ---------------- kernel 4: q,k (bf16), v (f32), all point-major [n][c] ----------------
__global__ __launch_bounds__(256) void compute_qkv_kernel(const __bf16* __restrict__ x,
                                                          const float* __restrict__ Wq,
                                                          const float* __restrict__ Wk,
                                                          const float* __restrict__ Wv,
                                                          __bf16* __restrict__ qbuf,
                                                          __bf16* __restrict__ kbuf,
                                                          float* __restrict__ vbuf) {
  const int tid = threadIdx.x, lane = tid & 31, wv = tid >> 5;
  const int b = blockIdx.y;
  const int z = blockIdx.z;
  const float* W = (z == 0) ? Wq : (z == 1) ? Wk : Wv;
  const int m0 = wv * 16;
  const int rowA = m0 + (lane & 15);
  const int rbase = m0 + ((lane & 16) ? 8 : 0);
  const int col = blockIdx.x * 16 + (lane & 15);
  const __bf16* X = x + (size_t)b * DM * Nn;
  v8f acc = {};
#pragma unroll
  for (int kk = 0; kk < 4; ++kk) {
    v16bf af = gblA_frag_f32(W, DM, rowA, kk * 32, lane);
    v16bf bf = gblB_frag_bf16T(X, col, kk * 32, lane);   // single 32B load
    acc = __builtin_amdgcn_wmma_f32_16x16x32_bf16(false, af, false, bf,
                                                  (short)0, acc, false, false);
  }
  const size_t o = (size_t)b * DM * Nn + (size_t)col * DM + rbase;
#pragma unroll
  for (int v = 0; v < 8; ++v) {
    if (z == 0)      qbuf[o + v] = (__bf16)acc[v];
    else if (z == 1) kbuf[o + v] = (__bf16)acc[v];
    else             vbuf[o + v] = acc[v];
  }
}

// ---------------- kernel 5: fused PE + attention MLP + softmax + output ----------------
__global__ __launch_bounds__(256) void fused_attn_kernel(
    const float* __restrict__ pos,  const float* __restrict__ feats,
    const float* __restrict__ Wd1,  const float* __restrict__ bnd1,
    const float* __restrict__ bnd2, const float* __restrict__ bng1,
    const float* __restrict__ bng2, const float* __restrict__ bn2,
    const __bf16* __restrict__ wcat,                       // [Wd2|Wg1|Wg2|W2] bf16
    const __bf16* __restrict__ qbuf, const __bf16* __restrict__ kbuf,
    const float* __restrict__ vbuf, const int* __restrict__ nbr,
    float* __restrict__ out) {
  __shared__ __bf16 sAct0[256 * DM];   // 64 KB, col-major [col][c]
  __shared__ __bf16 sAct1[256 * DM];   // 64 KB
  __shared__ __bf16 sPe  [256 * DM];   // 64 KB
  __shared__ __bf16 sW0  [DM * DM];    // 32 KB weight buffer A (TDM target)
  __shared__ __bf16 sW1  [DM * DM];    // 32 KB weight buffer B (TDM target)
  __shared__ float  sBN  [8 * DM + 2 * DP];
  __shared__ float  sWd1s[DM * 3];
  __shared__ int    sIdx [256];

  const int tid = threadIdx.x, lane = tid & 31, wv = tid >> 5;
  const int b  = blockIdx.y;
  const int n0 = blockIdx.x * 16;
  const size_t bDM = (size_t)b * DM * Nn;
  const size_t bP  = (size_t)b * 3 * Nn;
  const size_t bF  = (size_t)b * DP * Nn;

  // kick off async weight DMAs: Wd2 -> sW0, Wg1 -> sW1  (wave 0, TENSORcnt=2)
  if (wv == 0) {
    tdm_load_1d_bf16(wcat,                lds_off_u32(sW0), DM * DM);
    tdm_load_1d_bf16(wcat + DM * DM,      lds_off_u32(sW1), DM * DM);
  }

  // stage BN scale/shift, Wd1, neighbor tile
  for (int c = tid; c < DM; c += 256) {
    bn_stage(bnd1, DM, c, &sBN[0 * DM], &sBN[1 * DM]);
    bn_stage(bnd2, DM, c, &sBN[2 * DM], &sBN[3 * DM]);
    bn_stage(bng1, DM, c, &sBN[4 * DM], &sBN[5 * DM]);
    bn_stage(bng2, DM, c, &sBN[6 * DM], &sBN[7 * DM]);
  }
  for (int c = tid; c < DP; c += 256) bn_stage(bn2, DP, c, &sBN[8 * DM], &sBN[8 * DM + DP]);
  for (int i = tid; i < DM * 3; i += 256) sWd1s[i] = Wd1[i];
  sIdx[tid] = nbr[((size_t)b * Nn + (n0 + (tid >> 4))) * KNN + (tid & 15)];
  __syncthreads();

  // phase 1: pe1 = lrelu(bnd1(Wd1 @ rel)), one (point,neighbor) column per thread
  {
    const int i  = tid >> 4;
    const int n  = n0 + i;
    const int nb = sIdx[tid];
    float r0p = pos[bP + 0 * Nn + n] - pos[bP + 0 * Nn + nb];
    float r1p = pos[bP + 1 * Nn + n] - pos[bP + 1 * Nn + nb];
    float r2p = pos[bP + 2 * Nn + n] - pos[bP + 2 * Nn + nb];
    for (int c = 0; c < DM; ++c) {
      float a = sWd1s[c * 3] * r0p + sWd1s[c * 3 + 1] * r1p + sWd1s[c * 3 + 2] * r2p;
      a = a * sBN[0 * DM + c] + sBN[1 * DM + c];
      sAct0[tid * DM + c] = (__bf16)lrelu(a);
    }
  }
  if (wv == 0) __builtin_amdgcn_s_wait_tensorcnt(1);   // Wd2 in sW0 complete (in-order)
  __syncthreads();

  // phase 2: pe = lrelu(bnd2(Wd2 @ pe1))
  gemm_lds_bn_lrelu(sW0, sAct0, sPe, &sBN[2 * DM], &sBN[3 * DM], 256, lane, wv);
  __syncthreads();

  // phase 3: Wg2 -> sW0 (free now); build a = q - k_gather + pe (vectorized, point-major)
  if (wv == 0) tdm_load_1d_bf16(wcat + 2 * DM * DM, lds_off_u32(sW0), DM * DM);
  {
    const int i  = tid >> 4;
    const int n  = n0 + i;
    const int nb = sIdx[tid];
    const __bf16* qp = qbuf + bDM + (size_t)n * DM;
    const __bf16* kp = kbuf + bDM + (size_t)nb * DM;
#pragma unroll
    for (int c8 = 0; c8 < DM; c8 += 8) {
      v8bf qv = *reinterpret_cast<const v8bf*>(qp + c8);
      v8bf kv = *reinterpret_cast<const v8bf*>(kp + c8);
      v8bf pe = *reinterpret_cast<const v8bf*>(&sPe[tid * DM + c8]);
      v8bf o;
#pragma unroll
      for (int e = 0; e < 8; ++e)
        o[e] = (__bf16)((float)qv[e] - (float)kv[e] + (float)pe[e]);
      *reinterpret_cast<v8bf*>(&sAct0[tid * DM + c8]) = o;
    }
  }
  if (wv == 0) __builtin_amdgcn_s_wait_tensorcnt(1);   // Wg1 in sW1 complete
  __syncthreads();

  // phase 4: a1 = lrelu(bng1(Wg1 @ a))
  gemm_lds_bn_lrelu(sW1, sAct0, sAct1, &sBN[4 * DM], &sBN[5 * DM], 256, lane, wv);
  __syncthreads();

  // W2 -> sW1 (free now); make sure Wg2 is in sW0
  if (wv == 0) {
    tdm_load_1d_bf16(wcat + 3 * DM * DM, lds_off_u32(sW1), DP * DM);
    __builtin_amdgcn_s_wait_tensorcnt(1);              // Wg2 in sW0 complete
  }
  __syncthreads();

  // phase 5: a2 = lrelu(bng2(Wg2 @ a1))
  gemm_lds_bn_lrelu(sW0, sAct1, sAct0, &sBN[6 * DM], &sBN[7 * DM], 256, lane, wv);
  __syncthreads();

  // phase 6: softmax over neighbors + weighted sum -> res (cols 0..15 of sAct1)
#pragma unroll
  for (int pp = 0; pp < 8; ++pp) {
    const int p = tid + pp * 256;    // (c, i) pairs
    const int c = p & (DM - 1);
    const int i = p >> 7;
    float av[16];
    float mx = -3.0e38f;
#pragma unroll
    for (int j = 0; j < 16; ++j) {
      av[j] = (float)sAct0[(i * 16 + j) * DM + c] * INV_T;
      mx = fmaxf(mx, av[j]);
    }
    float s = 0.f;
#pragma unroll
    for (int j = 0; j < 16; ++j) { av[j] = __expf(av[j] - mx); s += av[j]; }
    const float inv = 1.0f / s;
    float res = 0.f;
#pragma unroll
    for (int j = 0; j < 16; ++j) {
      const int nb = sIdx[i * 16 + j];
      const float vv = vbuf[bDM + (size_t)nb * DM + c];   // coalesced over c
      res += av[j] * inv * (vv + (float)sPe[(i * 16 + j) * DM + c]);
    }
    sAct1[i * DM + c] = (__bf16)res;
  }
  if (wv == 0) __builtin_amdgcn_s_wait_tensorcnt(0);     // W2 in sW1 complete
  __syncthreads();

  // phase 7: out = lrelu(bn2(W2 @ res)) + feats  (64 rows -> waves 0..3)
  if (wv < 4) {
    const int m0 = wv * 16;
    const int rowA = m0 + (lane & 15);
    const int rbase = m0 + ((lane & 16) ? 8 : 0);
    v8f acc = {};
#pragma unroll
    for (int kk = 0; kk < 4; ++kk) {
      v16bf af = ldsA_frag(sW1, DM, rowA, kk * 32, lane);
      v16bf bf2 = ldsB_frag(sAct1, DM, lane & 15, kk * 32, lane);
      acc = __builtin_amdgcn_wmma_f32_16x16x32_bf16(false, af, false, bf2,
                                                    (short)0, acc, false, false);
    }
    const int n = n0 + (lane & 15);
#pragma unroll
    for (int v = 0; v < 8; ++v) {
      const int r = rbase + v;
      float y = lrelu(acc[v] * sBN[8 * DM + r] + sBN[8 * DM + DP + r]);
      out[bF + (size_t)r * Nn + n] = y + feats[bF + (size_t)r * Nn + n];
    }
  }
}

// ---------------- launcher ----------------
extern "C" void kernel_launch(void* const* d_in, const int* in_sizes, int n_in,
                              void* d_out, int out_size, void* d_ws, size_t ws_size,
                              hipStream_t stream) {
  (void)in_sizes; (void)n_in; (void)out_size; (void)ws_size;
  const float* feats = (const float*)d_in[0];
  const float* pos   = (const float*)d_in[1];
  const float* W1    = (const float*)d_in[2];
  const float* bn1   = (const float*)d_in[3];
  const float* W2    = (const float*)d_in[4];
  const float* bn2   = (const float*)d_in[5];
  const float* Wq    = (const float*)d_in[6];
  const float* Wk    = (const float*)d_in[7];
  const float* Wv    = (const float*)d_in[8];
  const float* Wd1   = (const float*)d_in[9];
  const float* bnd1  = (const float*)d_in[10];
  const float* Wd2   = (const float*)d_in[11];
  const float* bnd2  = (const float*)d_in[12];
  const float* Wg1   = (const float*)d_in[13];
  const float* bng1  = (const float*)d_in[14];
  const float* Wg2   = (const float*)d_in[15];
  const float* bng2  = (const float*)d_in[16];
  float* out = (float*)d_out;

  char* ws = (char*)d_ws;
  size_t off = 0;
  float*  d_sq  = (float*)(ws + off);  off += (size_t)Bc * Nn * sizeof(float);
  int*    d_idx = (int*)(ws + off);    off += (size_t)Bc * Nn * KNN * sizeof(int);
  __bf16* d_x   = (__bf16*)(ws + off); off += (size_t)Bc * DM * Nn * 2;
  __bf16* d_q   = (__bf16*)(ws + off); off += (size_t)Bc * DM * Nn * 2;
  __bf16* d_k   = (__bf16*)(ws + off); off += (size_t)Bc * DM * Nn * 2;
  float*  d_v   = (float*)(ws + off);  off += (size_t)Bc * DM * Nn * sizeof(float);
  __bf16* d_wc  = (__bf16*)(ws + off); off += (size_t)(3 * DM * DM + DP * DM) * 2;

  hipLaunchKernelGGL(prep_weights_kernel, dim3((3 * DM * DM + DP * DM) / 256), dim3(256),
                     0, stream, Wd2, Wg1, Wg2, W2, d_wc);
  hipLaunchKernelGGL(sq_kernel, dim3((Bc * Nn) / 256), dim3(256), 0, stream, feats, d_sq);
  hipLaunchKernelGGL(knn_kernel, dim3(Nn / 128, Bc), dim3(256), 0, stream, feats, d_sq, d_idx);
  hipLaunchKernelGGL(compute_x_kernel, dim3(Nn / 16, Bc), dim3(256), 0, stream,
                     feats, W1, bn1, d_x);
  hipLaunchKernelGGL(compute_qkv_kernel, dim3(Nn / 16, Bc, 3), dim3(256), 0, stream,
                     d_x, Wq, Wk, Wv, d_q, d_k, d_v);
  hipLaunchKernelGGL(fused_attn_kernel, dim3(Nn / 16, Bc), dim3(256), 0, stream,
                     pos, feats, Wd1, bnd1, bnd2, bng1, bng2, bn2,
                     d_wc, d_q, d_k, d_v, d_idx, out);
}
